// SPAIR_80668075753511
// MI455X (gfx1250) — compile-verified, matrix operand order checked by reading, not run on hardware
//
#include <hip/hip_runtime.h>
#include <math.h>

// ---------------------------------------------------------------------------
// Types
// ---------------------------------------------------------------------------
typedef __attribute__((ext_vector_type(16))) __bf16 v16bf;
typedef __attribute__((ext_vector_type(8)))  __bf16 v8bf;
typedef __attribute__((ext_vector_type(8)))  float  v8f;
typedef __attribute__((ext_vector_type(4)))  float  v4f;

#define BN   128   // batch
#define ACH  50    // attributes
#define NFX  100   // conv features
#define CTXW 56    // per-neighbor context width

__device__ __forceinline__ float sigmoidf_(float v){ return 1.f/(1.f+expf(-v)); }
__device__ __forceinline__ float softplusf_(float v){ return (v > 20.f) ? v : log1pf(expf(v)); }
__device__ __forceinline__ float csigf_(float v){ return fminf(fmaxf(sigmoidf_(v), 1e-6f), 1.f-1e-6f); }

// ---------------------------------------------------------------------------
// LDS region plan for the fused per-cell kernel (bytes, 256-aligned).
// Regions are time-multiplexed across the stage chain (lifetimes disjoint,
// separated by workgroup barriers). Peak = 300 KB < 320 KB/WGP.
//   R0: X1(90K) / X2(152K) / al.f32(57K) / ZI(120K) / OI(120K)
//   R1: h1 / e1(64K) / hz1 / o1
//   R2: hb / e2 / hz / o2          (32K)
//   R3: rep.f32 / dl.f32 / plog.f32 (8K)
//   R4: pth / pth2                 (28K)
//   R5: spf.f32 + boxb + attrb + depthb (16K)
// ---------------------------------------------------------------------------
#define OFF_R0 0
#define OFF_R1 155648
#define OFF_R2 (OFF_R1 + 65536)
#define OFF_R3 (OFF_R2 + 32768)
#define OFF_R4 (OFF_R3 + 8192)
#define OFF_R5 (OFF_R4 + 28672)
#define SMEM_TOTAL (OFF_R5 + 16384)   // 307200 bytes

// ---------------------------------------------------------------------------
// Weight packing: fp32 (K x N) [or (N x K) if transposed] -> bf16 fragments.
// For k-tile kt, n-tile nt, lane l: 16 contiguous halves = the exact v16bf
// B-fragment for V_WMMA_F32_16X16X32_BF16:
//   col n = nt*16 + (l&15), kbase = kt*32 + (l>=16 ? 8 : 0),
//   elem i<8 -> k = kbase+i, i>=8 -> k = kbase+8+i (+16..+23).
// Zero-padded beyond (K, N) so A-side zero padding contributes 0.
// ---------------------------------------------------------------------------
__global__ void pack_weight(const float* __restrict__ src, __bf16* __restrict__ dst,
                            int K, int N, int Kp, int transposed)
{
    int ntW = (N + 15) >> 4;
    int total = (Kp >> 5) * ntW * 512;
    int tid = blockIdx.x * blockDim.x + threadIdx.x;
    if (tid >= total) return;
    int i  = tid & 15;
    int l  = (tid >> 4) & 31;
    int fn = tid >> 9;
    int nt = fn % ntW;
    int kt = fn / ntW;
    int n = nt * 16 + (l & 15);
    int k = kt * 32 + ((l >= 16) ? 8 : 0) + (i < 8 ? i : i + 8);
    float v = 0.f;
    if (k < K && n < N)
        v = transposed ? src[(size_t)n * K + k] : src[(size_t)k * N + n];
    dst[tid] = (__bf16)v;
}

// ---------------------------------------------------------------------------
// Patch conv as implicit GEMM: M=32768 (b,oh,ow), N=100(pad112), K=768.
// A built on the fly from x(f32)->bf16 (8 consecutive k = 8 contiguous pixels).
// feat stored cell-major bf16: feat[(cell*128 + b)*100 + n], ReLU+bias.
// ---------------------------------------------------------------------------
__global__ void conv_wmma_kernel(const float* __restrict__ x, const __bf16* __restrict__ Wp,
                                 const float* __restrict__ bias, __bf16* __restrict__ feat)
{
    int lane = threadIdx.x & 31;
    int wv   = threadIdx.x >> 5;
    int mBase = blockIdx.x * 128 + wv * 16;

    int rA   = mBase + (lane & 15);
    int b    = rA >> 8;
    int cell = rA & 255;
    int oh   = cell >> 4;
    int ow   = cell & 15;
    int kb_off = (lane >= 16) ? 8 : 0;
    const int ntW = 7;

    for (int nt = 0; nt < ntW; ++nt) {
        v8f acc = {};
        for (int kt = 0; kt < 24; ++kt) {
            int kb0 = kt * 32 + kb_off;
            int kb1 = kb0 + 16;
            int c0 = kb0 >> 8, ky0 = (kb0 >> 4) & 15, kx0 = kb0 & 15;
            int c1 = kb1 >> 8, ky1 = (kb1 >> 4) & 15, kx1 = kb1 & 15;
            const float* p0 = x + (((size_t)(b * 3 + c0) * 256 + oh * 16 + ky0) * 256 + ow * 16 + kx0);
            const float* p1 = x + (((size_t)(b * 3 + c1) * 256 + oh * 16 + ky1) * 256 + ow * 16 + kx1);
            v4f f00 = *(const v4f*)p0;
            v4f f01 = *(const v4f*)(p0 + 4);
            v4f f10 = *(const v4f*)p1;
            v4f f11 = *(const v4f*)(p1 + 4);
            const v8bf* bp = (const v8bf*)(Wp + (((size_t)kt * ntW + nt) * 32 + lane) * 16);
            v8bf b0v = bp[0];
            v8bf b1v = bp[1];
            v16bf av, bv;
#pragma unroll
            for (int i = 0; i < 4; ++i) {
                av[i]      = (__bf16)f00[i];
                av[i + 4]  = (__bf16)f01[i];
                av[i + 8]  = (__bf16)f10[i];
                av[i + 12] = (__bf16)f11[i];
            }
#pragma unroll
            for (int i = 0; i < 8; ++i) { bv[i] = b0v[i]; bv[i + 8] = b1v[i]; }
            acc = __builtin_amdgcn_wmma_f32_16x16x32_bf16(false, av, false, bv,
                                                          (short)0, acc, false, false);
        }
        int n = nt * 16 + (lane & 15);
        if (n < NFX) {
            int rowAdd = (lane >= 16) ? 8 : 0;
#pragma unroll
            for (int j = 0; j < 8; ++j) {
                int m  = mBase + j + rowAdd;
                int bb = m >> 8;
                int cl = m & 255;
                float v = fmaxf(acc[j] + bias[n], 0.f);
                feat[((size_t)cl * BN + bb) * NFX + n] = (__bf16)v;
            }
        }
    }
}

// ---------------------------------------------------------------------------
// Context init: edge vector broadcast into ctx[b][17][18][56] (bf16).
// edge = [sigmoid(er[0:4]), er[4:54], er[54], sigmoid(er[55])]
// ---------------------------------------------------------------------------
__global__ void ctx_init(const float* __restrict__ er, __bf16* __restrict__ ctx)
{
    const int total = BN * 17 * 18 * CTXW;
    int tid = blockIdx.x * blockDim.x + threadIdx.x;
    if (tid >= total) return;
    int j = tid % CTXW;
    float v = er[j];
    if (j < 4 || j == 55) v = sigmoidf_(v);
    ctx[tid] = (__bf16)v;
}

// ---------------------------------------------------------------------------
// context = ctx[b][h][w..w+2][:] (168 contiguous) ++ ctx[b][h+1][w][:]
// ---------------------------------------------------------------------------
__device__ __forceinline__ __bf16 ctx_read(const __bf16* ctx, int b, int h, int w, int cc)
{
    if (cc < 168) return ctx[(((size_t)b * 17 + h) * 18 + w) * CTXW + cc];
    return ctx[(((size_t)b * 17 + h + 1) * 18 + w) * CTXW + (cc - 168)];
}

// ---------------------------------------------------------------------------
// In-workgroup GEMM stage: C[128 x N] = act(A[128 x Kp](LDS bf16) @ Wp + bias).
// 32 waves, tile (mt,nt): mt = 16-row batch tile, nt = 16-col tile of ldC.
// A fragments from LDS (ds_load_b128 x2), B fragments from packed global
// weights (global_load_b128 x2). Writes f32 and/or bf16 into LDS; cols >= N
// get 0 so the output doubles as a zero-padded A of the next GEMM.
// ---------------------------------------------------------------------------
__device__ __forceinline__ void gemm_stage(const __bf16* A, const __bf16* Wp,
                                           const float* bias, float* Cf, __bf16* Cb,
                                           int Kp, int N, int ldC, int relu)
{
    int tid  = threadIdx.x;
    int lane = tid & 31;
    int wv   = tid >> 5;
    int ntC  = ldC >> 4;
    int ntW  = (N + 15) >> 4;

    for (int t = wv; t < 8 * ntC; t += 32) {
        int mt = t & 7;
        int nt = t >> 3;
        v8f acc = {};
        if (nt < ntW) {
            int r = mt * 16 + (lane & 15);
            int kb_off = (lane >= 16) ? 8 : 0;
            const __bf16* arow = A + (size_t)r * Kp;
            int kTiles = Kp >> 5;
            for (int kt = 0; kt < kTiles; ++kt) {
                int kb = kt * 32 + kb_off;
                v8bf a0 = *(const v8bf*)(arow + kb);
                v8bf a1 = *(const v8bf*)(arow + kb + 16);
                const v8bf* bp = (const v8bf*)(Wp + (((size_t)kt * ntW + nt) * 32 + lane) * 16);
                v8bf b0 = bp[0];
                v8bf b1 = bp[1];
                v16bf av, bv;
#pragma unroll
                for (int i = 0; i < 8; ++i) {
                    av[i] = a0[i]; av[i + 8] = a1[i];
                    bv[i] = b0[i]; bv[i + 8] = b1[i];
                }
                acc = __builtin_amdgcn_wmma_f32_16x16x32_bf16(false, av, false, bv,
                                                              (short)0, acc, false, false);
            }
        }
        int col    = nt * 16 + (lane & 15);
        int rowAdd = (lane >= 16) ? 8 : 0;
#pragma unroll
        for (int j = 0; j < 8; ++j) {
            int r = mt * 16 + j + rowAdd;
            float v = 0.f;
            if (col < N) {
                v = acc[j] + bias[col];
                if (relu) v = fmaxf(v, 0.f);
            }
            if (Cf) Cf[(size_t)r * ldC + col] = v;
            if (Cb) Cb[(size_t)r * ldC + col] = (__bf16)v;
        }
    }
}

// ---------------------------------------------------------------------------
// Fused per-cell kernel: whole MLP chain for one grid cell, one workgroup
// (1024 threads / 32 waves) per cell, activations in LDS, stage barriers.
// Launched per anti-diagonal wavefront t = 2h + w (cells on one wavefront are
// independent); blockIdx.x selects h = hmin + blockIdx.x, w = t - 2h.
// ---------------------------------------------------------------------------
struct CellParams {
    const float* x;
    const __bf16* featB;
    __bf16* ctx;
    const __bf16 *p_bw1, *p_bw2, *p_bwo1, *p_bwo2;
    const __bf16 *p_ew1, *p_ew2, *p_ew3;
    const __bf16 *p_zw1, *p_zw2, *p_zwo1, *p_zwo2;
    const __bf16 *p_ow1, *p_ow2, *p_ow3;
    const float *bb1, *bb2, *bbo1, *bbo2;
    const float *eb1, *eb2, *eb3;
    const float *zb1, *zb2, *zbo1, *zbo2;
    const float *ob1, *ob2, *ob3;
    const float *eps_box, *eps_attr, *eps_depth, *u_pres;
    float* out;
};

__global__ __launch_bounds__(1024) void cell_step(CellParams P, int t, int hmin)
{
    extern __shared__ __align__(256) char smem[];
    int h = hmin + (int)blockIdx.x;
    int w = t - 2 * h;
    int cell = h * 16 + w;
    int tid = threadIdx.x;

    const __bf16* featCell = P.featB + (size_t)cell * BN * NFX;

    // region views (time-multiplexed; barriers separate lifetimes)
    __bf16* X1   = (__bf16*)(smem + OFF_R0);
    __bf16* X2   = (__bf16*)(smem + OFF_R0);
    float*  alf  = (float*) (smem + OFF_R0);
    __bf16* ZI   = (__bf16*)(smem + OFF_R0);
    __bf16* OI   = (__bf16*)(smem + OFF_R0);
    __bf16* h1   = (__bf16*)(smem + OFF_R1);
    __bf16* e1   = (__bf16*)(smem + OFF_R1);
    __bf16* hz1  = (__bf16*)(smem + OFF_R1);
    __bf16* o1   = (__bf16*)(smem + OFF_R1);
    __bf16* hb   = (__bf16*)(smem + OFF_R2);
    __bf16* e2   = (__bf16*)(smem + OFF_R2);
    __bf16* hz   = (__bf16*)(smem + OFF_R2);
    __bf16* o2   = (__bf16*)(smem + OFF_R2);
    float*  repf = (float*) (smem + OFF_R3);
    float*  dlf  = (float*) (smem + OFF_R3);
    float*  plf  = (float*) (smem + OFF_R3);
    __bf16* pth  = (__bf16*)(smem + OFF_R4);
    __bf16* pth2 = (__bf16*)(smem + OFF_R4);
    float*  spf  = (float*) (smem + OFF_R5);
    __bf16* boxb = (__bf16*)(smem + OFF_R5 + 2048);
    __bf16* attrb= (__bf16*)(smem + OFF_R5 + 3072);
    __bf16* depthb = (__bf16*)(smem + OFF_R5 + 15872);

    // warm weight lines for the first stages (global_prefetch_b8)
    if (tid < 512) {
        __builtin_prefetch(P.p_bw1 + (size_t)tid * 64, 0, 1);
        __builtin_prefetch(P.p_ew1 + (size_t)tid * 256, 0, 1);
    }

    // ---- stage 1: X1 = [feat | context], pad to 352 -------------------------
    for (int i = tid; i < BN * 352; i += 1024) {
        int b = i / 352, col = i % 352;
        __bf16 v = (__bf16)0.f;
        if (col < NFX)      v = featCell[(size_t)b * NFX + col];
        else if (col < 324) v = ctx_read(P.ctx, b, h, w, col - 100);
        X1[i] = v;
    }
    __syncthreads();

    gemm_stage(X1, P.p_bw1, P.bb1, nullptr, h1, 352, 100, 128, 1);
    __syncthreads();
    gemm_stage(h1, P.p_bw2, P.bb2, nullptr, hb, 128, 100, 128, 1);
    __syncthreads();
    gemm_stage(hb, P.p_bwo1, P.bbo1, repf, nullptr, 128, 8, 16, 0);
    gemm_stage(hb, P.p_bwo2, P.bbo2, nullptr, pth, 128, 100, 112, 0);
    __syncthreads();

    // ---- box head ----------------------------------------------------------
    if (tid < BN) {
        int b = tid;
        float lg[4];
#pragma unroll
        for (int j = 0; j < 4; ++j) {
            float m = repf[b * 16 + j];
            float s = softplusf_(repf[b * 16 + 4 + j]) + 1e-4f;
            lg[j] = m + s * P.eps_box[((size_t)(b * 4 + j) * 16 + h) * 16 + w];
        }
        float cy = 2.0f * csigf_(lg[0]) - 0.5f;
        float cx = 2.0f * csigf_(lg[1]) - 0.5f;
        float hh = 1.5f * csigf_(lg[2]) - 0.5f;
        float ww = 1.5f * csigf_(lg[3]) - 0.5f;
        float ysz = hh * (48.f / 256.f);
        float xsz = ww * (48.f / 256.f);
        float yt = (16.f / 256.f) * (cy + (float)h) - 0.5f * ysz;
        float xt = (16.f / 256.f) * (cx + (float)w) - 0.5f * xsz;
        P.out[((size_t)b * 57 + 0) * 256 + cell] = yt;
        P.out[((size_t)b * 57 + 1) * 256 + cell] = xt;
        P.out[((size_t)b * 57 + 2) * 256 + cell] = ysz;
        P.out[((size_t)b * 57 + 3) * 256 + cell] = xsz;
        spf[b * 4 + 0] = yt; spf[b * 4 + 1] = xt;
        spf[b * 4 + 2] = ysz; spf[b * 4 + 3] = xsz;
        boxb[b * 4 + 0] = (__bf16)cy; boxb[b * 4 + 1] = (__bf16)cx;
        boxb[b * 4 + 2] = (__bf16)hh; boxb[b * 4 + 3] = (__bf16)ww;
    }
    __syncthreads();

    // ---- STN bilinear glimpse into X2 (pad to 608) -------------------------
    for (int i = tid; i < BN * 608; i += 1024) {
        int b = i / 608, col = i % 608;
        __bf16 ov = (__bf16)0.f;
        if (col < 588) {
            int c = col / 196, p = col % 196, gy = p / 14, gx = p % 14;
            float yt = spf[b * 4 + 0], xt = spf[b * 4 + 1];
            float ys = spf[b * 4 + 2], xs = spf[b * 4 + 3];
            float gyv = -1.f + 2.f * (float)gy / 13.f;
            float gxv = -1.f + 2.f * (float)gx / 13.f;
            float ty = 2.f * (yt + 0.5f * ys) - 1.f;
            float tx = 2.f * (xt + 0.5f * xs) - 1.f;
            float py = (ys * gyv + ty + 1.f) * 0.5f * 255.f;
            float px = (xs * gxv + tx + 1.f) * 0.5f * 255.f;
            float y0 = floorf(py), x0 = floorf(px);
            float wy = py - y0, wx = px - x0;
            int y0i = (int)y0, x0i = (int)x0;
            const float* img = P.x + (size_t)(b * 3 + c) * 256 * 256;
            auto samp = [&](int yi, int xi) -> float {
                if (yi < 0 || yi > 255 || xi < 0 || xi > 255) return 0.f;
                return img[(size_t)yi * 256 + xi];
            };
            float v = (1.f - wy) * (1.f - wx) * samp(y0i, x0i)
                    + (1.f - wy) * wx        * samp(y0i, x0i + 1)
                    + wy        * (1.f - wx) * samp(y0i + 1, x0i)
                    + wy        * wx         * samp(y0i + 1, x0i + 1);
            ov = (__bf16)v;
        }
        X2[i] = ov;
    }
    __syncthreads();

    // ---- glimpse encoder ---------------------------------------------------
    gemm_stage(X2, P.p_ew1, P.eb1, nullptr, e1, 608, 256, 256, 1);
    __syncthreads();
    gemm_stage(e1, P.p_ew2, P.eb2, nullptr, e2, 256, 128, 128, 1);
    __syncthreads();
    gemm_stage(e2, P.p_ew3, P.eb3, alf, nullptr, 128, 100, 112, 0);
    __syncthreads();

    // ---- attribute head ----------------------------------------------------
    for (int i = tid; i < BN * ACH; i += 1024) {
        int b = i / ACH, j = i % ACH;
        float am = alf[(size_t)b * 112 + j];
        float as = softplusf_(alf[(size_t)b * 112 + ACH + j]) + 1e-4f;
        float at = am + as * P.eps_attr[((size_t)(b * ACH + j) * 16 + h) * 16 + w];
        P.out[((size_t)b * 57 + 4 + j) * 256 + cell] = at;
        attrb[b * ACH + j] = (__bf16)at;
    }
    __syncthreads();

    // ---- zi = [feat | ctx | passthru | box | attr], pad to 480 -------------
    for (int i = tid; i < BN * 480; i += 1024) {
        int b = i / 480, col = i % 480;
        __bf16 v = (__bf16)0.f;
        if (col < NFX)      v = featCell[(size_t)b * NFX + col];
        else if (col < 324) v = ctx_read(P.ctx, b, h, w, col - 100);
        else if (col < 424) v = pth[(size_t)b * 112 + (col - 324)];
        else if (col < 428) v = boxb[b * 4 + (col - 424)];
        else if (col < 478) v = attrb[b * ACH + (col - 428)];
        ZI[i] = v;
    }
    __syncthreads();

    gemm_stage(ZI, P.p_zw1, P.zb1, nullptr, hz1, 480, 100, 128, 1);
    __syncthreads();
    gemm_stage(hz1, P.p_zw2, P.zb2, nullptr, hz, 128, 100, 128, 1);
    __syncthreads();
    gemm_stage(hz, P.p_zwo1, P.zbo1, dlf, nullptr, 128, 2, 16, 0);
    gemm_stage(hz, P.p_zwo2, P.zbo2, nullptr, pth2, 128, 100, 112, 0);
    __syncthreads();

    // ---- depth head --------------------------------------------------------
    if (tid < BN) {
        int b = tid;
        float dm = dlf[b * 16 + 0];
        float ds = softplusf_(dlf[b * 16 + 1]) + 1e-4f;
        float depth = 4.f * csigf_(dm + ds * P.eps_depth[(size_t)b * 256 + cell]);
        P.out[((size_t)b * 57 + 54) * 256 + cell] = depth;
        depthb[b] = (__bf16)depth;
    }
    __syncthreads();

    // ---- oi = [feat | ctx | passthru2 | box | attr | depth], pad 480 -------
    for (int i = tid; i < BN * 480; i += 1024) {
        int b = i / 480, col = i % 480;
        __bf16 v = (__bf16)0.f;
        if (col < NFX)       v = featCell[(size_t)b * NFX + col];
        else if (col < 324)  v = ctx_read(P.ctx, b, h, w, col - 100);
        else if (col < 424)  v = pth2[(size_t)b * 112 + (col - 324)];
        else if (col < 428)  v = boxb[b * 4 + (col - 424)];
        else if (col < 478)  v = attrb[b * ACH + (col - 428)];
        else if (col == 478) v = depthb[b];
        OI[i] = v;
    }
    __syncthreads();

    gemm_stage(OI, P.p_ow1, P.ob1, nullptr, o1, 480, 100, 128, 1);
    __syncthreads();
    gemm_stage(o1, P.p_ow2, P.ob2, nullptr, o2, 128, 100, 128, 1);
    __syncthreads();
    gemm_stage(o2, P.p_ow3, P.ob3, plf, nullptr, 128, 1, 16, 0);
    __syncthreads();

    // ---- presence head + context writeback ---------------------------------
    if (tid < BN) {
        int b = tid;
        float pl = plf[b * 16];
        float prob = csigf_(pl);
        float u = P.u_pres[(size_t)b * 256 + cell];
        float pres = sigmoidf_(pl + logf(u) - log1pf(-u));
        P.out[((size_t)b * 57 + 55) * 256 + cell] = pres;
        P.out[((size_t)b * 57 + 56) * 256 + cell] = prob;
        __bf16* c = P.ctx + (((size_t)b * 17 + h + 1) * 18 + (w + 1)) * CTXW;
#pragma unroll
        for (int j = 0; j < 4; ++j) c[j] = boxb[b * 4 + j];
        for (int j = 0; j < ACH; ++j) c[4 + j] = attrb[b * ACH + j];
        c[54] = depthb[b];
        c[55] = (__bf16)pres;
    }
}

// ---------------------------------------------------------------------------
// Host orchestration
// ---------------------------------------------------------------------------
extern "C" void kernel_launch(void* const* d_in, const int* in_sizes, int n_in,
                              void* d_out, int out_size, void* d_ws, size_t ws_size,
                              hipStream_t stream)
{
    (void)in_sizes; (void)n_in; (void)out_size; (void)ws_size;

    const float* x        = (const float*)d_in[0];
    const float* conv_w   = (const float*)d_in[1];
    const float* conv_b   = (const float*)d_in[2];
    const float* edge_raw = (const float*)d_in[3];
    const float* bw1 = (const float*)d_in[4];  const float* bb1 = (const float*)d_in[5];
    const float* bw2 = (const float*)d_in[6];  const float* bb2 = (const float*)d_in[7];
    const float* bwo1= (const float*)d_in[8];  const float* bbo1= (const float*)d_in[9];
    const float* bwo2= (const float*)d_in[10]; const float* bbo2= (const float*)d_in[11];
    const float* ew1 = (const float*)d_in[12]; const float* eb1 = (const float*)d_in[13];
    const float* ew2 = (const float*)d_in[14]; const float* eb2 = (const float*)d_in[15];
    const float* ew3 = (const float*)d_in[16]; const float* eb3 = (const float*)d_in[17];
    const float* zw1 = (const float*)d_in[18]; const float* zb1 = (const float*)d_in[19];
    const float* zw2 = (const float*)d_in[20]; const float* zb2 = (const float*)d_in[21];
    const float* zwo1= (const float*)d_in[22]; const float* zbo1= (const float*)d_in[23];
    const float* zwo2= (const float*)d_in[24]; const float* zbo2= (const float*)d_in[25];
    const float* ow1 = (const float*)d_in[26]; const float* ob1 = (const float*)d_in[27];
    const float* ow2 = (const float*)d_in[28]; const float* ob2 = (const float*)d_in[29];
    const float* ow3 = (const float*)d_in[30]; const float* ob3 = (const float*)d_in[31];
    const float* eps_box   = (const float*)d_in[32];
    const float* eps_attr  = (const float*)d_in[33];
    const float* eps_depth = (const float*)d_in[34];
    const float* u_pres    = (const float*)d_in[35];
    float* out = (float*)d_out;

    // --- deterministic workspace allocator (256B aligned) ---
    size_t cur = 0;
    auto alloc = [&](size_t bytes) -> char* {
        char* p = (char*)d_ws + cur;
        cur += (bytes + 255) & ~(size_t)255;
        return p;
    };
    auto allocW = [&](int K, int N) -> __bf16* {
        int Kp = (K + 31) & ~31;
        int Np = ((N + 15) >> 4) << 4;
        return (__bf16*)alloc((size_t)Kp * Np * 2);
    };

    __bf16* featB = (__bf16*)alloc((size_t)256 * BN * NFX * 2);
    __bf16* ctxB  = (__bf16*)alloc((size_t)BN * 17 * 18 * CTXW * 2);

    __bf16* p_convw = allocW(768, 100);
    __bf16* p_bw1  = allocW(324, 100);  __bf16* p_bw2  = allocW(100, 100);
    __bf16* p_bwo1 = allocW(100, 8);    __bf16* p_bwo2 = allocW(100, 100);
    __bf16* p_ew1  = allocW(588, 256);  __bf16* p_ew2  = allocW(256, 128);
    __bf16* p_ew3  = allocW(128, 100);
    __bf16* p_zw1  = allocW(478, 100);  __bf16* p_zw2  = allocW(100, 100);
    __bf16* p_zwo1 = allocW(100, 2);    __bf16* p_zwo2 = allocW(100, 100);
    __bf16* p_ow1  = allocW(479, 100);  __bf16* p_ow2  = allocW(100, 100);
    __bf16* p_ow3  = allocW(100, 1);

    auto pack = [&](const float* src, __bf16* dst, int K, int N, int transposed) {
        int Kp = (K + 31) & ~31;
        int ntW = (N + 15) >> 4;
        int total = (Kp >> 5) * ntW * 512;
        pack_weight<<<(total + 255) / 256, 256, 0, stream>>>(src, dst, K, N, Kp, transposed);
    };

    // --- one-time setup: pack weights, init context, run conv ---
    pack(conv_w, p_convw, 768, 100, 1);
    pack(bw1,  p_bw1,  324, 100, 0);  pack(bw2,  p_bw2,  100, 100, 0);
    pack(bwo1, p_bwo1, 100,   8, 0);  pack(bwo2, p_bwo2, 100, 100, 0);
    pack(ew1,  p_ew1,  588, 256, 0);  pack(ew2,  p_ew2,  256, 128, 0);
    pack(ew3,  p_ew3,  128, 100, 0);
    pack(zw1,  p_zw1,  478, 100, 0);  pack(zw2,  p_zw2,  100, 100, 0);
    pack(zwo1, p_zwo1, 100,   2, 0);  pack(zwo2, p_zwo2, 100, 100, 0);
    pack(ow1,  p_ow1,  479, 100, 0);  pack(ow2,  p_ow2,  100, 100, 0);
    pack(ow3,  p_ow3,  100,   1, 0);

    {
        int total = BN * 17 * 18 * CTXW;
        ctx_init<<<(total + 255) / 256, 256, 0, stream>>>(edge_raw, ctxB);
    }
    conv_wmma_kernel<<<256, 256, 0, stream>>>(x, p_convw, conv_b, featB);

    // --- fused per-cell chain over anti-diagonal wavefronts t = 2h + w ------
    CellParams P;
    P.x = x; P.featB = featB; P.ctx = ctxB;
    P.p_bw1 = p_bw1; P.p_bw2 = p_bw2; P.p_bwo1 = p_bwo1; P.p_bwo2 = p_bwo2;
    P.p_ew1 = p_ew1; P.p_ew2 = p_ew2; P.p_ew3 = p_ew3;
    P.p_zw1 = p_zw1; P.p_zw2 = p_zw2; P.p_zwo1 = p_zwo1; P.p_zwo2 = p_zwo2;
    P.p_ow1 = p_ow1; P.p_ow2 = p_ow2; P.p_ow3 = p_ow3;
    P.bb1 = bb1; P.bb2 = bb2; P.bbo1 = bbo1; P.bbo2 = bbo2;
    P.eb1 = eb1; P.eb2 = eb2; P.eb3 = eb3;
    P.zb1 = zb1; P.zb2 = zb2; P.zbo1 = zbo1; P.zbo2 = zbo2;
    P.ob1 = ob1; P.ob2 = ob2; P.ob3 = ob3;
    P.eps_box = eps_box; P.eps_attr = eps_attr; P.eps_depth = eps_depth;
    P.u_pres = u_pres; P.out = out;

    for (int t = 0; t <= 45; ++t) {
        int hmin = (t > 15) ? ((t - 15 + 1) >> 1) : 0;
        int hmax = (t >> 1) < 15 ? (t >> 1) : 15;
        int nb = hmax - hmin + 1;
        cell_step<<<nb, 1024, SMEM_TOTAL, stream>>>(P, t, hmin);
    }
}
